// LLaMALayer_48430051229868
// MI455X (gfx1250) — matrix operand order repaired and checked by reading, not compiled, not run on hardware
//
#include <hip/hip_runtime.h>
#include <stdint.h>

typedef __attribute__((ext_vector_type(16))) __bf16 v16bf;
typedef __attribute__((ext_vector_type(8)))  float  v8f;
typedef __attribute__((ext_vector_type(4))) unsigned int u32x4;
typedef __attribute__((ext_vector_type(8))) int i32x8;
typedef __attribute__((ext_vector_type(4))) int i32x4;

#define RMS_EPS 1.1920929e-07f

// ---- WMMA fragment K-index maps (per CDNA5 ISA 7.12.2) ----
// A (16x32, 16-bit): VGPR v, lane-half h -> K base
__device__ __forceinline__ int kmapA(int v, int h) {
    return ((v < 4) ? (2 * v) : (16 + 2 * (v - 4))) + 8 * h;
}
// B (32x16, 16-bit): same K order as A (lanes 0-15: K=0..7,16..23; lanes 16-31: K=8..15,24..31)
__device__ __forceinline__ int kmapB(int v, int h) {
    return ((v < 4) ? (2 * v) : (16 + 2 * (v - 4))) + 8 * h;
}

__device__ __forceinline__ v8f v8f_zero() {
    v8f z;
#pragma unroll
    for (int i = 0; i < 8; i++) z[i] = 0.0f;
    return z;
}

// ============================================================
// Tensor Data Mover: 2D tile load global->LDS (D# per ISA 08_async_tensor §8).
// lds_byte_addr = LDS offset (low 32 bits of flat LDS pointer).
// pad codes: interval 0=2,1=4,2=8,3=16,4=32,5=64 dwords; amount 0=1..7=8 (code = dwords-1).
// ============================================================
__device__ __forceinline__ void tdm_load_2d(uint32_t lds_byte_addr, const void* gptr,
                                            uint32_t tensor_d0, uint32_t tensor_d1,
                                            uint32_t tile_d0, uint32_t tile_d1,
                                            uint32_t stride_d0,
                                            uint32_t pad_interval_code,
                                            uint32_t pad_amount_code,
                                            uint32_t pad_enable) {
    uint64_t ga = (uint64_t)(uintptr_t)gptr;
    u32x4 g0;
    g0[0] = 1u;                                              // count=1 (valid descriptor)
    g0[1] = lds_byte_addr;                                   // lds_addr
    g0[2] = (uint32_t)ga;                                    // global_addr[31:0]
    g0[3] = (uint32_t)((ga >> 32) & 0x1FFFFFFu) | (2u << 30); // global_addr[56:32] | type=2
    i32x8 g1;
    g1[0] = (int)((1u << 16) |                               // data_size = 2 bytes (bf16)
                  (pad_enable << 20) |
                  (pad_interval_code << 22) |
                  (pad_amount_code << 25));
    g1[1] = (int)((tensor_d0 & 0xFFFFu) << 16);              // tensor_dim0[15:0]
    g1[2] = (int)(((tensor_d0 >> 16) & 0xFFFFu) |
                  ((tensor_d1 & 0xFFFFu) << 16));            // tensor_dim0[31:16] | tensor_dim1[15:0]
    g1[3] = (int)(((tensor_d1 >> 16) & 0xFFFFu) |
                  ((tile_d0 & 0xFFFFu) << 16));              // tensor_dim1[31:16] | tile_dim0
    g1[4] = (int)(tile_d1 & 0xFFFFu);                        // tile_dim1 (tile_dim2=0)
    g1[5] = (int)stride_d0;                                  // tensor_dim0_stride[31:0]
    g1[6] = 0;
    g1[7] = 0;
    i32x4 z4 = {0, 0, 0, 0};
#if __clang_major__ >= 23
    i32x8 z8 = {0, 0, 0, 0, 0, 0, 0, 0};
    __builtin_amdgcn_tensor_load_to_lds(g0, g1, z4, z4, z8, 0);
#else
    __builtin_amdgcn_tensor_load_to_lds(g0, g1, z4, z4, 0);
#endif
}

// ============================================================
// RMSNorm: fp32 in -> bf16 out (one block per row)
// ============================================================
__global__ __launch_bounds__(256) void rmsnorm_kernel(const float* __restrict__ x,
                                                      const float* __restrict__ g,
                                                      __bf16* __restrict__ out,
                                                      int cols) {
    int row = blockIdx.x;
    const float* xr = x + (size_t)row * cols;
    float ss = 0.0f;
    for (int c = threadIdx.x; c < cols; c += 256) { float v = xr[c]; ss += v * v; }
#pragma unroll
    for (int off = 16; off > 0; off >>= 1) ss += __shfl_xor(ss, off, 32);
    __shared__ float red[8];
    int wave = threadIdx.x >> 5;
    if ((threadIdx.x & 31) == 0) red[wave] = ss;
    __syncthreads();
    float tot = 0.0f;
#pragma unroll
    for (int i = 0; i < 8; i++) tot += red[i];
    float rinv = rsqrtf(tot / (float)cols + RMS_EPS);
    __bf16* orow = out + (size_t)row * cols;
    for (int c = threadIdx.x; c < cols; c += 256) orow[c] = (__bf16)(xr[c] * rinv * g[c]);
}

// ============================================================
// GEMM: C[M,N] = A[M,K](bf16, row-major) * W[K,N](fp32, row-major)
// A tile staged via TDM (async tensor DMA, padded LDS rows);
// fp32 weights converted to bf16 while staged into LDS (read once from HBM).
// Block tile 64x128, 8 waves, each wave: 16 rows x 64 cols = 4 WMMA accumulators.
// ============================================================
template <int OUT_BF16>
__global__ __launch_bounds__(256) void gemm_kernel(const __bf16* __restrict__ A,
                                                   const float* __restrict__ W,
                                                   void* __restrict__ Cv,
                                                   int M, int N, int K) {
    __shared__ __bf16 sA[64][48];    // [m][k]: 32 elem rows (16 dw) + 8 dw pad = 48 halves
    __shared__ __bf16 sB[128][48];   // [n][k] (transposed weight tile), padded
    int tid = threadIdx.x;
    int lane = tid & 31, wave = tid >> 5;
    int half = lane >> 4, l16 = lane & 15;
    int wm = wave & 3, wn = wave >> 2;
    int bm = blockIdx.x * 64, bn = blockIdx.y * 128;

    v8f acc[4];
#pragma unroll
    for (int j = 0; j < 4; j++) acc[j] = v8f_zero();

    for (int kt = 0; kt < K; kt += 32) {
        // stage A tile 64x32 via Tensor Data Mover (wave 0 issues, TENSORcnt tracks)
        if (wave == 0) {
            tdm_load_2d((uint32_t)(uintptr_t)&sA[0][0],
                        A + (size_t)bm * K + kt,
                        (uint32_t)K, (uint32_t)(M - bm),
                        /*tile*/ 32, 64, (uint32_t)K,
                        /*pad: 16 dw interval*/ 3, /*8 dw amount*/ 7, 1);
        }
        // prefetch next weight panel while converting this one
        if (kt + 32 < K)
            __builtin_prefetch(W + (size_t)(kt + 32) * N + bn + (tid & 127), 0, 1);
        // stage W tile: 32x128 fp32 -> bf16, stored transposed [n][k]
        {
            int n = tid & 127;
            int ks = (tid >> 7) << 4;
            const float* wp = W + (size_t)(kt + ks) * N + bn + n;
#pragma unroll
            for (int i = 0; i < 16; i++) sB[n][ks + i] = (__bf16)wp[(size_t)i * N];
        }
        if (wave == 0) __builtin_amdgcn_s_wait_tensorcnt(0);
        __syncthreads();

        v16bf af;
        int arow = wm * 16 + l16;
#pragma unroll
        for (int v = 0; v < 8; v++) {
            int k = kmapA(v, half);
            af[2 * v]     = sA[arow][k];
            af[2 * v + 1] = sA[arow][k + 1];
        }
#pragma unroll
        for (int j = 0; j < 4; j++) {
            v16bf bfr;
            int bcol = wn * 64 + j * 16 + l16;
#pragma unroll
            for (int v = 0; v < 8; v++) {
                int k = kmapB(v, half);
                bfr[2 * v]     = sB[bcol][k];
                bfr[2 * v + 1] = sB[bcol][k + 1];
            }
            acc[j] = __builtin_amdgcn_wmma_f32_16x16x32_bf16(
                false, af, false, bfr, (short)0, acc[j], false, false);
        }
        __syncthreads();
    }

#pragma unroll
    for (int j = 0; j < 4; j++) {
#pragma unroll
        for (int r = 0; r < 8; r++) {
            int row = bm + wm * 16 + r + 8 * half;
            int col = bn + wn * 64 + j * 16 + l16;
            if (OUT_BF16)
                ((__bf16*)Cv)[(size_t)row * N + col] = (__bf16)acc[j][r];
            else
                ((float*)Cv)[(size_t)row * N + col] = acc[j][r];
        }
    }
}

// ============================================================
// RoPE (neox-style split halves), bf16 -> bf16, out of place.
// ============================================================
__global__ __launch_bounds__(256) void rope_kernel(const __bf16* __restrict__ in,
                                                   __bf16* __restrict__ out) {
    int idx = blockIdx.x * 256 + threadIdx.x;  // S*H*64 total
    int i = idx & 63;
    int hh = (idx >> 6) & 31;
    int s = idx >> 11;
    const __bf16* ip = in + (size_t)s * 4096 + hh * 128;
    __bf16* op = out + (size_t)s * 4096 + hh * 128;
    float xe = (float)ip[2 * i], xo = (float)ip[2 * i + 1];
    float freq = __expf(-(float)(2 * i) * (9.2103403719761836f / 128.0f)); // 10000^(-2i/128)
    float ang = (float)s * freq;
    float c = __cosf(ang), sn = __sinf(ang);
    op[i]      = (__bf16)(xe * c - xo * sn);
    op[64 + i] = (__bf16)(xe * sn + xo * c);
}

// ============================================================
// Flash attention: block = (1 head, 128 q-rows), 8 waves x 16 rows.
// K tiles staged by TDM; V transposed manually; scores never touch HBM.
// ============================================================
__global__ __launch_bounds__(256) void flash_attn_kernel(const __bf16* __restrict__ Q,
                                                         const __bf16* __restrict__ Kb,
                                                         const __bf16* __restrict__ Vb,
                                                         __bf16* __restrict__ O,
                                                         int S) {
    __shared__ __bf16 sK[32][136];       // [key][hd]: 128 elem (64 dw) + 4 dw pad
    __shared__ __bf16 sV[128][48];       // [hd][key] (transposed for B-frag reads)
    __shared__ __bf16 sP[8][16][48];     // per-wave probs [row][key]
    int tid = threadIdx.x;
    int lane = tid & 31, wave = tid >> 5;
    int half = lane >> 4, l16 = lane & 15;
    int h = blockIdx.y;
    int qr0 = blockIdx.x * 128 + wave * 16;

    // Q fragments: head_dim 128 -> 4 A-frags, loaded once from global
    v16bf qf[4];
    const __bf16* qp = Q + (size_t)(qr0 + l16) * 4096 + h * 128;
#pragma unroll
    for (int f = 0; f < 4; f++) {
#pragma unroll
        for (int v = 0; v < 8; v++) {
            int k = f * 32 + kmapA(v, half);
            qf[f][2 * v]     = qp[k];
            qf[f][2 * v + 1] = qp[k + 1];
        }
    }

    v8f o[8];
#pragma unroll
    for (int j = 0; j < 8; j++) o[j] = v8f_zero();
    float m[8], l[8];
#pragma unroll
    for (int r = 0; r < 8; r++) { m[r] = -3.0e38f; l[r] = 0.0f; }

    for (int kb = 0; kb < S; kb += 32) {
        __syncthreads();
        // stage K block [32][128] via Tensor Data Mover
        if (wave == 0) {
            tdm_load_2d((uint32_t)(uintptr_t)&sK[0][0],
                        Kb + (size_t)kb * 4096 + h * 128,
                        4096, (uint32_t)(S - kb),
                        /*tile*/ 128, 32, 4096,
                        /*pad: 64 dw interval*/ 5, /*4 dw amount*/ 3, 1);
        }
        // stage V block transposed -> sV[hd][key]
        {
            int key = tid & 31;
            int hd0 = (tid >> 5) << 4;
            const __bf16* vp = Vb + (size_t)(kb + key) * 4096 + h * 128 + hd0;
#pragma unroll
            for (int i = 0; i < 16; i++) sV[hd0 + i][key] = vp[i];
        }
        if (wave == 0) __builtin_amdgcn_s_wait_tensorcnt(0);
        __syncthreads();

        // scores: (16x128) x (128x32) -> two 16x16 C tiles
        v8f s0 = v8f_zero(), s1 = v8f_zero();
#pragma unroll
        for (int f = 0; f < 4; f++) {
            v16bf kf0, kf1;
#pragma unroll
            for (int v = 0; v < 8; v++) {
                int k = f * 32 + kmapB(v, half);
                kf0[2 * v]     = sK[l16][k];
                kf0[2 * v + 1] = sK[l16][k + 1];
                kf1[2 * v]     = sK[16 + l16][k];
                kf1[2 * v + 1] = sK[16 + l16][k + 1];
            }
            s0 = __builtin_amdgcn_wmma_f32_16x16x32_bf16(false, qf[f], false, kf0, (short)0, s0, false, false);
            s1 = __builtin_amdgcn_wmma_f32_16x16x32_bf16(false, qf[f], false, kf1, (short)0, s1, false, false);
        }

        // online softmax (row reductions across 16-lane groups via shfl_xor)
#pragma unroll
        for (int r = 0; r < 8; r++) {
            float v0 = s0[r], v1 = s1[r];
            float mx = fmaxf(v0, v1);
#pragma unroll
            for (int off = 8; off > 0; off >>= 1) mx = fmaxf(mx, __shfl_xor(mx, off, 32));
            float mnew = fmaxf(m[r], mx);
            float p0 = __expf(v0 - mnew);
            float p1 = __expf(v1 - mnew);
            float rs = p0 + p1;
#pragma unroll
            for (int off = 8; off > 0; off >>= 1) rs += __shfl_xor(rs, off, 32);
            float sc = __expf(m[r] - mnew);
            l[r] = l[r] * sc + rs;
            m[r] = mnew;
#pragma unroll
            for (int j = 0; j < 8; j++) o[j][r] *= sc;
            int prow = r + 8 * half;
            sP[wave][prow][l16]      = (__bf16)p0;
            sP[wave][prow][16 + l16] = (__bf16)p1;
        }
        // intra-wave LDS turnaround (C-layout -> A-frag re-layout through sP)
        __builtin_amdgcn_wave_barrier();
        asm volatile("s_wait_dscnt 0" ::: "memory");

        // O += P(16x32) @ V(32x128)
        v16bf pf;
#pragma unroll
        for (int v = 0; v < 8; v++) {
            int k = kmapA(v, half);
            pf[2 * v]     = sP[wave][l16][k];
            pf[2 * v + 1] = sP[wave][l16][k + 1];
        }
#pragma unroll
        for (int j = 0; j < 8; j++) {
            v16bf vf;
#pragma unroll
            for (int v = 0; v < 8; v++) {
                int k = kmapB(v, half);
                vf[2 * v]     = sV[j * 16 + l16][k];
                vf[2 * v + 1] = sV[j * 16 + l16][k + 1];
            }
            o[j] = __builtin_amdgcn_wmma_f32_16x16x32_bf16(false, pf, false, vf, (short)0, o[j], false, false);
        }
    }

    // finalize: divide by row sums, write [s][head][hd] (== [s][4096] layout)
#pragma unroll
    for (int r = 0; r < 8; r++) {
        float inv = 1.0f / l[r];
        int row = qr0 + r + 8 * half;
#pragma unroll
        for (int j = 0; j < 8; j++) {
            O[(size_t)row * 4096 + h * 128 + j * 16 + l16] = (__bf16)(o[j][r] * inv);
        }
    }
}

// ============================================================
// gated = x1 * silu(x3), fp32 in -> bf16 out
// ============================================================
__global__ __launch_bounds__(256) void silu_mul_kernel(const float* __restrict__ x1,
                                                       const float* __restrict__ x3,
                                                       __bf16* __restrict__ out, int n) {
    int idx = blockIdx.x * 256 + threadIdx.x;
    if (idx < n) {
        float a = x1[idx], b = x3[idx];
        float s = b / (1.0f + __expf(-b));
        out[idx] = (__bf16)(a * s);
    }
}

// ============================================================
extern "C" void kernel_launch(void* const* d_in, const int* in_sizes, int n_in,
                              void* d_out, int out_size, void* d_ws, size_t ws_size,
                              hipStream_t stream) {
    (void)in_sizes; (void)n_in; (void)out_size; (void)ws_size;
    const float* x      = (const float*)d_in[0];
    const float* wq     = (const float*)d_in[1];
    const float* wk     = (const float*)d_in[2];
    const float* wv     = (const float*)d_in[3];
    const float* wo     = (const float*)d_in[4];
    const float* w1     = (const float*)d_in[5];
    const float* w2     = (const float*)d_in[6];
    const float* w3     = (const float*)d_in[7];
    const float* g_attn = (const float*)d_in[8];
    const float* g_ffn  = (const float*)d_in[9];

    const int S = 2048, D = 4096, F = 11008, H = 32;

    char* ws = (char*)d_ws;
    size_t off = 0;
    auto alloc = [&](size_t bytes) -> char* {
        char* p = ws + off;
        off += (bytes + 255) & ~(size_t)255;
        return p;
    };
    __bf16* hbf  = (__bf16*)alloc((size_t)S * D * 2);
    __bf16* tmp  = (__bf16*)alloc((size_t)S * D * 2);
    __bf16* qbf  = (__bf16*)alloc((size_t)S * D * 2);
    __bf16* kbf  = (__bf16*)alloc((size_t)S * D * 2);
    __bf16* vbf  = (__bf16*)alloc((size_t)S * D * 2);
    __bf16* abf  = (__bf16*)alloc((size_t)S * D * 2);
    float*  xo   = (float*)alloc((size_t)S * D * 4);
    __bf16* h2bf = (__bf16*)alloc((size_t)S * D * 2);
    float*  x1f  = (float*)alloc((size_t)S * F * 4);
    float*  x3f  = (float*)alloc((size_t)S * F * 4);
    __bf16* gbf  = (__bf16*)alloc((size_t)S * F * 2);

    dim3 blk(256);
    dim3 gD(S / 64, D / 128);   // 32 x 32
    dim3 gF(S / 64, F / 128);   // 32 x 86

    // 1) h = rmsnorm(x, g_attn)   (fp32 -> bf16)
    rmsnorm_kernel<<<S, blk, 0, stream>>>(x, g_attn, hbf, D);
    // 2) q/k/v projections + RoPE
    gemm_kernel<1><<<gD, blk, 0, stream>>>(hbf, wq, tmp, S, D, D);
    rope_kernel<<<(S * H * 64) / 256, blk, 0, stream>>>(tmp, qbf);
    gemm_kernel<1><<<gD, blk, 0, stream>>>(hbf, wk, tmp, S, D, D);
    rope_kernel<<<(S * H * 64) / 256, blk, 0, stream>>>(tmp, kbf);
    gemm_kernel<1><<<gD, blk, 0, stream>>>(hbf, wv, vbf, S, D, D);
    // 3) attention (fused flash: softmax(QK^T)V, scores stay on-chip)
    flash_attn_kernel<<<dim3(S / 128, H), blk, 0, stream>>>(qbf, kbf, vbf, abf, S);
    // 4) output projection (fp32 out; feeds rmsnorm)
    gemm_kernel<0><<<gD, blk, 0, stream>>>(abf, wo, xo, S, D, D);
    // 5) h2 = rmsnorm(xo, g_ffn)
    rmsnorm_kernel<<<S, blk, 0, stream>>>(xo, g_ffn, h2bf, D);
    // 6) FFN up projections
    gemm_kernel<0><<<gF, blk, 0, stream>>>(h2bf, w1, x1f, S, F, D);
    gemm_kernel<0><<<gF, blk, 0, stream>>>(h2bf, w3, x3f, S, F, D);
    // 7) gated = x1 * silu(x3)
    int n = S * F;
    silu_mul_kernel<<<(n + 255) / 256, blk, 0, stream>>>(x1f, x3f, gbf, n);
    // 8) out = gated @ w2   (fp32 straight to d_out)
    gemm_kernel<0><<<gD, blk, 0, stream>>>(gbf, w2, (float*)d_out, S, D, F);
}
